// SDEBlock_83262236000597
// MI455X (gfx1250) — matrix operand order repaired and checked by reading, not compiled, not run on hardware
//
#include <hip/hip_runtime.h>
#include <hip/hip_bf16.h>
#include <stdint.h>

#define B_ROWS  131072
#define DDIM    128
#define N_STEPS 10
#define TILE_M  128   // rows per block (8 waves x 16 rows)

typedef _Float16     v16h __attribute__((ext_vector_type(16)));
typedef float        v8f  __attribute__((ext_vector_type(8)));
typedef unsigned int v4u  __attribute__((ext_vector_type(4)));
typedef int          v8i  __attribute__((ext_vector_type(8)));
typedef int          v4i  __attribute__((ext_vector_type(4)));

__device__ __forceinline__ float fast_tanh(float x) {
#if __has_builtin(__builtin_amdgcn_tanhf)
    return __builtin_amdgcn_tanhf(x);
#else
    float e = __builtin_amdgcn_exp2f(x * 2.8853900817779268f);
    return 1.0f - 2.0f * __builtin_amdgcn_rcpf(e + 1.0f);
#endif
}

__device__ __forceinline__ float nt_load(const float* p) {
#if __has_builtin(__builtin_nontemporal_load)
    return __builtin_nontemporal_load(p);
#else
    return *p;
#endif
}

#if __has_builtin(__builtin_amdgcn_tensor_load_to_lds) && \
    __has_builtin(__builtin_amdgcn_tensor_store_from_lds)
#define HAVE_TDM 1
#else
#define HAVE_TDM 0
#endif

#if HAVE_TDM
// 2D TDM descriptor for a 128x128 tile of f32, row stride 128 elements.
__device__ __forceinline__ void tdm_tile_128x128_f32(bool is_store,
                                                     unsigned lds_off,
                                                     const void* gptr) {
    unsigned long long ga = (unsigned long long)(uintptr_t)gptr;
    v4u g0;
    g0[0] = 1u;                                              // count=1, user mode
    g0[1] = lds_off;                                         // lds_addr (bytes)
    g0[2] = (unsigned)ga;                                    // global_addr[31:0]
    g0[3] = (unsigned)((ga >> 32) & 0x01FFFFFFull)
          | (2u << 30);                                      // addr[56:32] | type=2
    v8i g1;
    g1[0] = (2 << 16);        // data_size = 4 bytes; wg_mask=0 (not in cluster)
    g1[1] = (128 << 16);      // tensor_dim0 = 128
    g1[2] = (128 << 16);      // tensor_dim1 = 128 (low half)
    g1[3] = (128 << 16);      // tensor_dim1 hi = 0 | tile_dim0 = 128
    g1[4] = 128;              // tile_dim1 = 128, tile_dim2 = 0
    g1[5] = 128;              // tensor_dim0_stride = 128
    g1[6] = 0;
    g1[7] = 0;
    v4i gz = {0, 0, 0, 0};
#if defined(__clang_major__) && (__clang_major__ >= 23)
    v8i gz8 = {0, 0, 0, 0, 0, 0, 0, 0};
    if (is_store) __builtin_amdgcn_tensor_store_from_lds(g0, g1, gz, gz, gz8, 0);
    else          __builtin_amdgcn_tensor_load_to_lds   (g0, g1, gz, gz, gz8, 0);
#else
    if (is_store) __builtin_amdgcn_tensor_store_from_lds(g0, g1, gz, gz, 0);
    else          __builtin_amdgcn_tensor_load_to_lds   (g0, g1, gz, gz, 0);
#endif
}
#endif  // HAVE_TDM

__global__ __launch_bounds__(256) void sde_mlp_em_kernel(
    const float* __restrict__ x,  const float* __restrict__ W1,
    const float* __restrict__ b1, const float* __restrict__ W2,
    const float* __restrict__ b2, const float* __restrict__ noise,
    float* __restrict__ out)
{
    // 160 KB LDS total
    __shared__ __align__(32) float    xs [TILE_M][DDIM];  // 64 KB running state (fp32)
    __shared__ __align__(32) _Float16 w1t[DDIM][DDIM];    // 32 KB W1^T (f16)
    __shared__ __align__(32) _Float16 w2t[DDIM][DDIM];    // 32 KB W2^T (f16)
    __shared__ __align__(32) _Float16 hs [TILE_M][DDIM];  // 32 KB hidden (f16)

    const int tid   = threadIdx.x;
    const int wave  = tid >> 5;
    const int lane  = tid & 31;
    const int l16   = lane & 15;
    const int lhalf = lane >> 4;        // 0: lanes 0-15, 1: lanes 16-31
    const int m0    = wave * 16;        // this wave's 16 rows in the tile
    const int r0    = blockIdx.x * TILE_M;

    // ---- stage x tile into LDS (TDM DMA; overlaps weight conversion) ----
#if HAVE_TDM
    if (wave == 0)
        tdm_tile_128x128_f32(false, (unsigned)(uintptr_t)&xs[0][0],
                             x + (size_t)r0 * DDIM);
#else
    for (int idx = tid; idx < TILE_M * DDIM; idx += 256)
        xs[idx >> 7][idx & 127] = x[(size_t)r0 * DDIM + idx];
#endif

    // ---- stage transposed f16 weights (reused 20 GEMM passes) ----
    for (int idx = tid; idx < DDIM * DDIM; idx += 256) {
        int k = idx >> 7, n = idx & 127;
        float a = W1[idx], bb = W2[idx];
        w1t[n][k] = (_Float16)a;        // w1t[n][k] = W1[k][n]
        w2t[n][k] = (_Float16)bb;
    }

    // ---- biases in registers (per-lane column set) ----
    float b1r[8], b2r[8];
#pragma unroll
    for (int nt = 0; nt < 8; ++nt) {
        b1r[nt] = b1[nt * 16 + l16];
        b2r[nt] = b2[nt * 16 + l16];
    }

#if HAVE_TDM
    if (wave == 0) __builtin_amdgcn_s_wait_tensorcnt(0);
#endif
    __syncthreads();

    const float DT  = 0.1f;                        // dt
    const float SDT = 0.1f * 0.31622776601683794f; // sigma * sqrt(dt)
    const int   arow = m0 + l16;                   // A-fragment row for this lane
    const int   crow = m0 + 8 * lhalf;             // C-layout base row

#pragma unroll 1
    for (int s = 0; s < N_STEPS; ++s) {
        __syncthreads();  // convergent; blocks LICM from hoisting B-frags

        // ---- A1 fragments: f32 x -> f16, contiguous k order per lane half ----
        // half j (0..15) holds K = kc*32 + 16*lhalf + j
        v16h a1[4];
#pragma unroll
        for (int kc = 0; kc < 4; ++kc) {
            int kb = kc * 32 + 16 * lhalf;
#pragma unroll
            for (int j = 0; j < 8; ++j) {
                a1[kc][j]     = (_Float16)xs[arow][kb + j];
                a1[kc][j + 8] = (_Float16)xs[arow][kb + 8 + j];
            }
        }

        // ---- GEMM1: h = tanh(x @ W1 + b1), B-frags double-buffered ----
        {
            v16h bcur[4], bnxt[4];
#pragma unroll
            for (int kc = 0; kc < 4; ++kc)
                bcur[kc] = *(const v16h*)&w1t[l16][kc * 32 + 16 * lhalf];
#pragma unroll
            for (int nt = 0; nt < 8; ++nt) {
                int n = nt * 16 + l16;
                if (nt < 7) {
#pragma unroll
                    for (int kc = 0; kc < 4; ++kc)
                        bnxt[kc] = *(const v16h*)&w1t[n + 16][kc * 32 + 16 * lhalf];
                }
                v8f c = {0.f, 0.f, 0.f, 0.f, 0.f, 0.f, 0.f, 0.f};
#pragma unroll
                for (int kc = 0; kc < 4; ++kc)
                    c = __builtin_amdgcn_wmma_f32_16x16x32_f16(
                            false, a1[kc], false, bcur[kc], (short)0, c,
                            false, false);
                // batched tanh (fills TRANS hazard slots), then scatter
                float t[8];
#pragma unroll
                for (int i = 0; i < 8; ++i) t[i] = fast_tanh(c[i] + b1r[nt]);
#pragma unroll
                for (int i = 0; i < 8; ++i)
                    hs[crow + i][n] = (_Float16)t[i];
#pragma unroll
                for (int kc = 0; kc < 4; ++kc) bcur[kc] = bnxt[kc];
            }
        }
        // hs rows are wave-private: same-wave DS ordering suffices

        // ---- A2 fragments from f16 h ----
        v16h a2[4];
#pragma unroll
        for (int kc = 0; kc < 4; ++kc) {
            int kb = kc * 32 + 16 * lhalf;
#pragma unroll
            for (int j = 0; j < 8; ++j) {
                a2[kc][j]     = hs[arow][kb + j];
                a2[kc][j + 8] = hs[arow][kb + 8 + j];
            }
        }

        // ---- GEMM2 + Euler-Maruyama update (noise streamed, non-temporal) ----
        const float* zbase = noise + ((size_t)s * B_ROWS + r0) * DDIM;
        {
            v16h bcur[4], bnxt[4];
#pragma unroll
            for (int kc = 0; kc < 4; ++kc)
                bcur[kc] = *(const v16h*)&w2t[l16][kc * 32 + 16 * lhalf];
#pragma unroll
            for (int nt = 0; nt < 8; ++nt) {
                int n = nt * 16 + l16;
                // issue the tile's noise loads first: HBM latency hides
                // under the WMMAs below
                float z[8];
#pragma unroll
                for (int i = 0; i < 8; ++i)
                    z[i] = nt_load(&zbase[(size_t)(crow + i) * DDIM + n]);
                if (nt < 7) {
#pragma unroll
                    for (int kc = 0; kc < 4; ++kc)
                        bnxt[kc] = *(const v16h*)&w2t[n + 16][kc * 32 + 16 * lhalf];
                }
                v8f c = {0.f, 0.f, 0.f, 0.f, 0.f, 0.f, 0.f, 0.f};
#pragma unroll
                for (int kc = 0; kc < 4; ++kc)
                    c = __builtin_amdgcn_wmma_f32_16x16x32_f16(
                            false, a2[kc], false, bcur[kc], (short)0, c,
                            false, false);
#pragma unroll
                for (int i = 0; i < 8; ++i) {
                    int   m  = crow + i;
                    float f  = c[i] + b2r[nt];
                    float xo = xs[m][n];
                    // x_next = x*(1 + sigma*sqrt(dt)*z) + f*dt
                    xs[m][n] = fmaf(f, DT, xo * fmaf(SDT, z[i], 1.0f));
                }
#pragma unroll
                for (int kc = 0; kc < 4; ++kc) bcur[kc] = bnxt[kc];
            }
        }
    }

    __syncthreads();

    // ---- write the tile back (TDM DMA from LDS) ----
#if HAVE_TDM
    if (wave == 0) {
        tdm_tile_128x128_f32(true, (unsigned)(uintptr_t)&xs[0][0],
                             out + (size_t)r0 * DDIM);
        __builtin_amdgcn_s_wait_tensorcnt(0);
    }
#else
    for (int idx = tid; idx < TILE_M * DDIM; idx += 256)
        out[(size_t)r0 * DDIM + idx] = xs[idx >> 7][idx & 127];
#endif
}

extern "C" void kernel_launch(void* const* d_in, const int* in_sizes, int n_in,
                              void* d_out, int out_size, void* d_ws, size_t ws_size,
                              hipStream_t stream) {
    (void)in_sizes; (void)n_in; (void)out_size; (void)d_ws; (void)ws_size;
    const float* x     = (const float*)d_in[0];
    const float* W1    = (const float*)d_in[1];
    const float* b1    = (const float*)d_in[2];
    const float* W2    = (const float*)d_in[3];
    const float* b2    = (const float*)d_in[4];
    const float* noise = (const float*)d_in[5];
    float*       out   = (float*)d_out;

    dim3 grid(B_ROWS / TILE_M);   // 1024 blocks
    dim3 block(256);              // 8 wave32s
    sde_mlp_em_kernel<<<grid, block, 0, stream>>>(x, W1, b1, W2, b2, noise, out);
}
